// VITSelfAttention_89996744721127
// MI455X (gfx1250) — matrix operand order repaired and checked by reading, not compiled, not run on hardware
//
#include <hip/hip_runtime.h>

// =====================================================================
// VIT channel self-attention for MI455X (gfx1250, wave32, WMMA).
//
// Phase 1: fused Q/K/V projection + sigmoid via v_wmma_f32_16x16x32_bf16.
//          All three weight matrices staged in LDS (96 KB of the 320 KB
//          WGP budget) so the input X is read once and A-fragments are
//          reused across the three projections. Outputs stored as bf16
//          in [B, C, Hp, Wp] layout with packed b128 stores (memory-
//          bound problem: ~17 GFLOP vs ~300 MB at 23.3 TB/s).
// Phase 2: per-(b,c) channel attention. Tiles pulled into LDS with
//          GLOBAL_LOAD_ASYNC_TO_LDS_B128 (ASYNCcnt), then pool, two
//          scaled WMMA GEMM + softmax stages, two more WMMA GEMMs,
//          writing [B, N, C] f32 output.
// =====================================================================

typedef __bf16 bf16_t;
typedef __attribute__((ext_vector_type(16))) __bf16 v16bf;
typedef __attribute__((ext_vector_type(8)))  __bf16 v8bf;
typedef __attribute__((ext_vector_type(8)))  float  v8f;
typedef __attribute__((ext_vector_type(4)))  float  v4f;

constexpr int B_  = 64;
constexpr int N_  = 2048;
constexpr int C_  = 128;
constexpr int HP_ = 64;
constexpr int WP_ = 32;
constexpr int G_  = 32;            // HP / POOL
constexpr float D_SCALE = 0.08838834764831843f;  // 1/sqrt(128)

// ---------------------------------------------------------------------
// Fragment gather helpers (ISA 7.12.2 layouts, wave32).
// A 16x32 bf16: lane half selects K-base 0/8; two runs {b..b+7},{b+16..b+23}
// B 32x16 bf16 stored as arr[n][k] (K contiguous): same K order as A, b=0/8
// ---------------------------------------------------------------------
__device__ __forceinline__ v16bf frag_A(const bf16_t* __restrict__ row, int lane) {
  const int base = (lane < 16) ? 0 : 8;
  v8bf c0 = *(const v8bf*)(row + base);
  v8bf c1 = *(const v8bf*)(row + base + 16);
  v16bf a;
#pragma unroll
  for (int i = 0; i < 8; ++i) { a[i] = c0[i]; a[8 + i] = c1[i]; }
  return a;
}

__device__ __forceinline__ v16bf frag_B(const bf16_t* __restrict__ row, int lane) {
  const int base = (lane < 16) ? 0 : 8;
  v8bf c0 = *(const v8bf*)(row + base);
  v8bf c1 = *(const v8bf*)(row + base + 16);
  v16bf b;
#pragma unroll
  for (int i = 0; i < 8; ++i) { b[i] = c0[i]; b[8 + i] = c1[i]; }
  return b;
}

__device__ __forceinline__ v8f wmma_bf16(v16bf a, v16bf b, v8f c) {
  return __builtin_amdgcn_wmma_f32_16x16x32_bf16(false, a, false, b,
                                                 (short)0, c, false, false);
}

// Async 16-byte global -> LDS copy (CDNA5 GLOBAL_LOAD_ASYNC_TO_LDS_B128,
// GV mode, tracked by ASYNCcnt). LDS operand is the low-32-bit LDS byte
// address of the generic pointer (ISA 10.2: LDS aperture addr[31:0]).
__device__ __forceinline__ void async_copy_b128(void* lds_ptr, const void* gptr) {
  const unsigned lds_off = (unsigned)(unsigned long long)lds_ptr;
  asm volatile("global_load_async_to_lds_b128 %0, %1, off"
               :: "v"(lds_off), "v"((unsigned long long)gptr)
               : "memory");
}

__device__ __forceinline__ void wait_asynccnt0() {
  asm volatile("s_wait_asynccnt 0x0" ::: "memory");
}

// =====================================================================
// Phase 1: sigmoid(X @ W^T + b) for Q,K,V -> bf16 spatial workspace.
// grid = 1024, block = 256 (8 waves). Each wave: 16 rows x 128 cols
// per projection, K=128 as 4 WMMA chunks per N-tile, 3 projections.
// =====================================================================
__global__ __launch_bounds__(256) void qkv_proj_kernel(
    const float* __restrict__ X,
    const float* __restrict__ Wq, const float* __restrict__ bq,
    const float* __restrict__ Wk, const float* __restrict__ bk,
    const float* __restrict__ Wv, const float* __restrict__ bv,
    bf16_t* __restrict__ ws)
{
  __shared__ __align__(16) bf16_t Wlds[3][C_ * C_];  // W[n][k], K contiguous (96 KB)
  __shared__ __align__(16) float  biasL[3 * C_];

  const size_t plane = (size_t)B_ * C_ * N_;
  const int tid = threadIdx.x;

  {
    const float* Ws[3] = {Wq, Wk, Wv};
    const float* Bs[3] = {bq, bk, bv};
    for (int p = 0; p < 3; ++p)
      for (int i = tid; i < C_ * C_; i += 256) Wlds[p][i] = (bf16_t)Ws[p][i];
    for (int i = tid; i < 3 * C_; i += 256) biasL[i] = Bs[i >> 7][i & 127];
  }
  __syncthreads();

  const int lane = tid & 31;
  const int wave = tid >> 5;
  const int rowBase = (blockIdx.x * 8 + wave) * 16;
  const int nl   = lane & 15;
  const int moff = (lane < 16) ? 0 : 8;

  // Preload the wave's A fragments once: 16 rows x 128 K as 4 chunks.
  // Each 8-float run is contiguous & 32B aligned -> b128 vector loads.
  v16bf afrag[4];
  {
    const float* row = X + (size_t)(rowBase + nl) * C_;
    const int kb = (lane < 16) ? 0 : 8;
#pragma unroll
    for (int kc = 0; kc < 4; ++kc) {
      const float* p = row + kc * 32 + kb;
      v4f f0 = *(const v4f*)(p);
      v4f f1 = *(const v4f*)(p + 4);
      v4f f2 = *(const v4f*)(p + 16);
      v4f f3 = *(const v4f*)(p + 20);
      v16bf a;
#pragma unroll
      for (int i = 0; i < 4; ++i) {
        a[i]      = (bf16_t)f0[i];
        a[4 + i]  = (bf16_t)f1[i];
        a[8 + i]  = (bf16_t)f2[i];
        a[12 + i] = (bf16_t)f3[i];
      }
      afrag[kc] = a;
    }
  }

  // Per-lane packed-store base: the 8 accumulator rows are consecutive
  // nseq positions of the same (b, n) plane -> one b128 store per tile.
  const int r0    = rowBase + moff;
  const int bidx  = r0 >> 11;        // / 2048 (constant across the 8 rows)
  const int nseq0 = r0 & 2047;

  for (int p = 0; p < 3; ++p) {
    bf16_t* dst = ws + (size_t)p * plane;
    const bf16_t* Wp = &Wlds[p][0];
#pragma unroll
    for (int nt = 0; nt < 8; ++nt) {
      const int n = nt * 16 + nl;
      const float bval = biasL[p * C_ + n];
      v8f acc;
#pragma unroll
      for (int j = 0; j < 8; ++j) acc[j] = bval;  // bias (same n for all 8 rows)
#pragma unroll
      for (int kc = 0; kc < 4; ++kc) {
        v16bf bfr = frag_B(Wp + n * C_ + kc * 32, lane);
        acc = wmma_bf16(afrag[kc], bfr, acc);
      }
      // sigmoid + packed 16B store into spatial layout dst[(b*C+n)*N + nseq0..+7]
      v8bf pk;
#pragma unroll
      for (int j = 0; j < 8; ++j)
        pk[j] = (bf16_t)(1.0f / (1.0f + __expf(-acc[j])));
      *(v8bf*)(dst + ((size_t)bidx * C_ + n) * N_ + nseq0) = pk;
    }
  }
}

// =====================================================================
// Phase 2: per-(b,c) channel attention. grid = B*C = 8192, block = 128.
// =====================================================================
__global__ __launch_bounds__(128) void chan_attn_kernel(
    const bf16_t* __restrict__ ws, float* __restrict__ out)
{
  __shared__ __align__(16) float  S1f[HP_ * G_];    // logits Qk [h][g]
  __shared__ __align__(16) float  S2f[G_ * HP_];    // logits Kq [g][h]
  __shared__ __align__(16) bf16_t Qbf [HP_ * WP_];  // Q[h][w]   (async-filled)
  __shared__ __align__(16) bf16_t Kbf [HP_ * WP_];  // K[h][w]   (async-filled)
  __shared__ __align__(16) bf16_t Vlin[HP_ * WP_];  // V[h][w]   (async-filled)
  __shared__ __align__(16) bf16_t VT  [WP_ * HP_];  // V^T[w][h] (B for Kqv)
  __shared__ __align__(16) bf16_t qbf [G_  * WP_];  // pooled q[g][w]
  __shared__ __align__(16) bf16_t kpbf[G_  * WP_];  // pooled k[g][w]
  __shared__ __align__(16) bf16_t P1bf[HP_ * G_];   // softmax(Qk)[h][g]
  __shared__ __align__(16) bf16_t P2bf[G_  * HP_];  // softmax(Kq)[g][h]
  __shared__ __align__(16) bf16_t KqvT[WP_ * G_];   // Kqv^T[w][g]

  const int tid  = threadIdx.x;
  const int pair = blockIdx.x;                       // b*C + c
  const size_t plane = (size_t)B_ * C_ * N_;
  const size_t off   = (size_t)pair * (HP_ * WP_);
  const bf16_t* Qg = ws + off;
  const bf16_t* Kg = ws + plane + off;
  const bf16_t* Vg = ws + 2 * plane + off;

  // ---- Async copy the three 4 KB tiles straight into LDS ----
  for (int i = tid; i < 256; i += 128) {             // 256 x 16B per tile
    const int bo = i * 16;
    async_copy_b128((char*)Qbf  + bo, (const char*)Qg + bo);
    async_copy_b128((char*)Kbf  + bo, (const char*)Kg + bo);
    async_copy_b128((char*)Vlin + bo, (const char*)Vg + bo);
  }
  wait_asynccnt0();
  __syncthreads();

  // ---- Transpose V (LDS->LDS) and pool Q,K over h (window 2) ----
  for (int i = tid; i < HP_ * WP_; i += 128) {
    const int h = i >> 5, w = i & 31;
    VT[w * HP_ + h] = Vlin[i];
  }
  for (int i = tid; i < G_ * WP_; i += 128) {
    const int g = i >> 5, w = i & 31;
    qbf[i]  = (bf16_t)(0.5f * ((float)Qbf[(2 * g) * WP_ + w] +
                               (float)Qbf[(2 * g + 1) * WP_ + w]));
    kpbf[i] = (bf16_t)(0.5f * ((float)Kbf[(2 * g) * WP_ + w] +
                               (float)Kbf[(2 * g + 1) * WP_ + w]));
  }
  __syncthreads();

  const int lane = tid & 31;
  const int wv   = tid >> 5;
  const int nl   = lane & 15;
  const int moff = (lane < 16) ? 0 : 8;

  // ---- Logit GEMMs: S1 = scale*Q@k^T (64x32), S2 = scale*q@K^T (32x64) ----
#pragma unroll
  for (int rep = 0; rep < 2; ++rep) {
    const int t = wv + rep * 4;
    {   // S1 tile: mt 0..3 over h, nt 0..1 over g, K = w (32)
      const int mt = t >> 1, nt = t & 1;
      v16bf a = frag_A(Qbf  + (mt * 16 + nl) * WP_, lane);
      v16bf b = frag_B(kpbf + (nt * 16 + nl) * WP_, lane);
      v8f acc = {};
      acc = wmma_bf16(a, b, acc);
#pragma unroll
      for (int j = 0; j < 8; ++j)
        S1f[(mt * 16 + moff + j) * G_ + nt * 16 + nl] = D_SCALE * acc[j];
    }
    {   // S2 tile: mt 0..1 over g, nt 0..3 over h, K = w (32)
      const int mt = t >> 2, nt = t & 3;
      v16bf a = frag_A(qbf + (mt * 16 + nl) * WP_, lane);
      v16bf b = frag_B(Kbf + (nt * 16 + nl) * WP_, lane);
      v8f acc = {};
      acc = wmma_bf16(a, b, acc);
#pragma unroll
      for (int j = 0; j < 8; ++j)
        S2f[(mt * 16 + moff + j) * HP_ + nt * 16 + nl] = D_SCALE * acc[j];
    }
  }
  __syncthreads();

  // ---- Row softmaxes: S1 over g (64 rows x 32), S2 over h (32 rows x 64) ----
  if (tid < 64) {
    const int h = tid;
    const float* row = S1f + h * G_;
    float mx = row[0];
    for (int g = 1; g < G_; ++g) mx = fmaxf(mx, row[g]);
    float sum = 0.0f;
    for (int g = 0; g < G_; ++g) sum += __expf(row[g] - mx);
    const float inv = 1.0f / sum;
    for (int g = 0; g < G_; ++g)
      P1bf[h * G_ + g] = (bf16_t)(__expf(row[g] - mx) * inv);
  } else if (tid < 96) {
    const int g = tid - 64;
    const float* row = S2f + g * HP_;
    float mx = row[0];
    for (int h = 1; h < HP_; ++h) mx = fmaxf(mx, row[h]);
    float sum = 0.0f;
    for (int h = 0; h < HP_; ++h) sum += __expf(row[h] - mx);
    const float inv = 1.0f / sum;
    for (int h = 0; h < HP_; ++h)
      P2bf[g * HP_ + h] = (bf16_t)(__expf(row[h] - mx) * inv);
  }
  __syncthreads();

  // ---- Kqv = P2 @ V (32x32), K = h (64, two chunks); one tile per wave ----
  {
    const int mt = wv >> 1, nt = wv & 1;
    v8f acc = {};
#pragma unroll
    for (int kc = 0; kc < 2; ++kc) {
      v16bf a = frag_A(P2bf + (mt * 16 + nl) * HP_ + kc * 32, lane);
      v16bf b = frag_B(VT   + (nt * 16 + nl) * HP_ + kc * 32, lane);
      acc = wmma_bf16(a, b, acc);
    }
    // 8 consecutive g positions -> one packed 16B ds store
    v8bf pk;
#pragma unroll
    for (int j = 0; j < 8; ++j) pk[j] = (bf16_t)acc[j];
    *(v8bf*)(&KqvT[(nt * 16 + nl) * G_ + mt * 16 + moff]) = pk;
  }
  __syncthreads();

  // ---- out = P1 @ Kqv (64x32), K = g (32); write [B, N, C] f32 ----
  const int bidx = pair >> 7;        // / 128
  const int c    = pair & 127;
#pragma unroll
  for (int rep = 0; rep < 2; ++rep) {
    const int t  = wv + rep * 4;
    const int mt = t >> 1, nt = t & 1;
    v16bf a = frag_A(P1bf + (mt * 16 + nl) * G_, lane);
    v16bf b = frag_B(KqvT + (nt * 16 + nl) * G_, lane);
    v8f acc = {};
    acc = wmma_bf16(a, b, acc);
    const int w = nt * 16 + nl;
#pragma unroll
    for (int j = 0; j < 8; ++j) {
      const int h = mt * 16 + moff + j;
      out[((size_t)bidx * N_ + h * WP_ + w) * C_ + c] = acc[j];
    }
  }
}

// =====================================================================
extern "C" void kernel_launch(void* const* d_in, const int* in_sizes, int n_in,
                              void* d_out, int out_size, void* d_ws, size_t ws_size,
                              hipStream_t stream) {
  const float* X  = (const float*)d_in[0];
  const float* Wq = (const float*)d_in[1];
  const float* bq = (const float*)d_in[2];
  const float* Wk = (const float*)d_in[3];
  const float* bk = (const float*)d_in[4];
  const float* Wv = (const float*)d_in[5];
  const float* bv = (const float*)d_in[6];
  bf16_t* ws = (bf16_t*)d_ws;   // needs 3 * 64*128*2048 * 2B = 96 MiB
  float* out = (float*)d_out;

  qkv_proj_kernel<<<1024, 256, 0, stream>>>(X, Wq, bq, Wk, bk, Wv, bv, ws);
  chan_attn_kernel<<<B_ * C_, 128, 0, stream>>>(ws, out);
}
